// K_MOTE_4449586119086
// MI455X (gfx1250) — matrix operand order repaired and checked by reading, not compiled, not run on hardware
//
#include <hip/hip_runtime.h>
#include <hip/hip_bf16.h>

typedef __attribute__((ext_vector_type(16))) _Float16 v16h;
typedef __attribute__((ext_vector_type(8)))  float    v8f;

#define THREADS 256
#define NWAVES  8
#define NBLOCKS 512

// LDS holds all 4 expert weight matrices, zero-padded to K=32, pre-swizzled
// into the v_wmma f16 B-matrix (32x16 per N-tile) lane layout:
//   dword index = ((e*8 + nt)*32 + lane)*8 + j
//   lane   -> column N = lane&15 ; K-half = lane>>4 (lanes 0-15: K 0-7,16-23; 16-31: K 8-15,24-31)
//   dword j -> packed {K = k0 (lo16), K = k0+1 (hi16)}, k0 = 8*(lane>>4) + 2j + 8*(j>>2)
#define LDS_DWORDS (4 * 8 * 32 * 8) // 8192 dwords = 32 KB

__device__ __forceinline__ void run_expert(v16h a, int e, int lane,
                                           const unsigned int* ldsB,
                                           size_t rowBase,
                                           float* __restrict__ emb)
{
  const int m  = lane & 15;
  const int hi = lane >> 4;
#pragma unroll
  for (int nt = 0; nt < 8; ++nt) {
    const v16h bv = *(const v16h*)&ldsB[(size_t)(((e * 8 + nt) * 32) + lane) * 8];
    v8f c = {};
    // D = A(16x32 f16) x B(32x16 f16) + 0  -> v_wmma_f32_16x16x32_f16
    c = __builtin_amdgcn_wmma_f32_16x16x32_f16(
        /*neg_a=*/false, a, /*neg_b=*/false, bv,
        /*c_mod=*/(short)0, c, /*reuse_a=*/false, /*reuse_b=*/false);
    // D layout: VGPR r, lane l -> row M = r + 8*(l>=16), col N = l&15
    size_t base = rowBase + (size_t)(8 * hi) * 512 + (size_t)(e * 128 + nt * 16 + m);
#pragma unroll
    for (int r = 0; r < 8; ++r)
      emb[base + (size_t)r * 512] = c[r];
  }
}

__global__ __launch_bounds__(THREADS) void mote_moe_kernel(
    const float* __restrict__ ts,       const float* __restrict__ aux,
    const float* __restrict__ rW,       const float* __restrict__ rb,
    const float* __restrict__ freqs,    const float* __restrict__ fourier_W,
    const float* __restrict__ knots,    const float* __restrict__ spline_W,
    const float* __restrict__ centers,  const float* __restrict__ gauss_W,
    const float* __restrict__ wav_c,    const float* __restrict__ wav_s,
    const float* __restrict__ wavelet_W,
    float* __restrict__ emb, float* __restrict__ raw_out, float* __restrict__ msk_out,
    int nRows)
{
  __shared__ __align__(32) unsigned int ldsB[LDS_DWORDS];

  const int tid = threadIdx.x;

  // ---- one-time per block: pack f32 weights -> f16 B-tiles in LDS ----
  // tid + i*256 makes the tile index (idx>>8) == i, so e/nt fold to constants.
#pragma unroll
  for (int i = 0; i < LDS_DWORDS / THREADS; ++i) {
    int idx  = tid + i * THREADS;
    int j    = idx & 7;
    int lane = (idx >> 3) & 31;
    int tl   = idx >> 8;                   // == i (compile-time)
    int e    = tl >> 3;
    int nt   = tl & 7;
    int col  = nt * 16 + (lane & 15);
    int k0   = ((lane >> 4) << 3) + 2 * j + ((j >> 2) << 3); // even K of the packed pair
    const float* W = (e == 0) ? fourier_W
                   : (e == 1) ? spline_W
                   : (e == 2) ? gauss_W
                              : wavelet_W;
    float w0 = 0.0f, w1 = 0.0f;
    if (e == 0) {                 // fourier: full K=32
      w0 = W[k0 * 128 + col];
      w1 = W[(k0 + 1) * 128 + col];
    } else if (k0 < 16) {         // K=16 experts: zero-pad upper half
      w0 = W[k0 * 128 + col];
      w1 = W[(k0 + 1) * 128 + col];
    }
    union { _Float16 h[2]; unsigned int u; } pk;
    pk.h[0] = (_Float16)w0; pk.h[1] = (_Float16)w1;
    ldsB[idx] = pk.u;
  }
  __syncthreads();

  const int lane   = tid & 31;
  const int waveId = tid >> 5;
  const int gWave  = blockIdx.x * NWAVES + waveId;
  const int totW   = gridDim.x * NWAVES;
  const int m      = lane & 15;     // row within tile (A/M, also N for stores)
  const int hi     = lane >> 4;     // K-half selector for A layout

  // ---- per-lane invariant basis params (A 16x32 layout: slot j -> K=hi*8+j,
  //      slot j+8 -> K=16+hi*8+j; fourier cos reuses freqs[hi*8+j]) ----
  float pf_freq[8], pf_knot[8], pf_cent[8], pf_wc[8], pf_wsi[8];
#pragma unroll
  for (int j = 0; j < 8; ++j) {
    int p      = hi * 8 + j;
    pf_freq[j] = freqs[p];
    pf_knot[j] = knots[p];
    pf_cent[j] = centers[p];
    pf_wc[j]   = wav_c[p];
    pf_wsi[j]  = 1.0f / wav_s[p];
  }
  const float inv_h = 15.0f / 4.0f;   // 1/h, h = 4/(G-1), G=16

  const int nTiles = nRows >> 4;
  for (int tile = gWave; tile < nTiles; tile += totW) {
    const int    row     = tile * 16 + m;          // both lane halves: same row
    const size_t rowBase = (size_t)tile * 16 * 512;
    const float  tv      = ts[row];

    // ---- router: logits = [t, aux] @ rW + rb (uniform weights -> s_loads) ----
    float lg0 = rb[0] + tv * rW[0];
    float lg1 = rb[1] + tv * rW[1];
    float lg2 = rb[2] + tv * rW[2];
    float lg3 = rb[3] + tv * rW[3];
    const float4* a4 = (const float4*)(aux + (size_t)row * 16);
    float av[16];
    {
      float4 x0 = a4[0], x1 = a4[1], x2 = a4[2], x3 = a4[3];
      av[0]=x0.x; av[1]=x0.y; av[2]=x0.z; av[3]=x0.w;
      av[4]=x1.x; av[5]=x1.y; av[6]=x1.z; av[7]=x1.w;
      av[8]=x2.x; av[9]=x2.y; av[10]=x2.z; av[11]=x2.w;
      av[12]=x3.x; av[13]=x3.y; av[14]=x3.z; av[15]=x3.w;
    }
#pragma unroll
    for (int i = 0; i < 16; ++i) {
      lg0 += av[i] * rW[(1 + i) * 4 + 0];
      lg1 += av[i] * rW[(1 + i) * 4 + 1];
      lg2 += av[i] * rW[(1 + i) * 4 + 2];
      lg3 += av[i] * rW[(1 + i) * 4 + 3];
    }
    // softmax
    float mx = fmaxf(fmaxf(lg0, lg1), fmaxf(lg2, lg3));
    float e0 = __expf(lg0 - mx), e1 = __expf(lg1 - mx);
    float e2 = __expf(lg2 - mx), e3 = __expf(lg3 - mx);
    float inv = 1.0f / (e0 + e1 + e2 + e3);
    float w0 = e0 * inv, w1 = e1 * inv, w2 = e2 * inv, w3 = e3 * inv;
    // top-2 (ties -> lower index, matching lax.top_k)
    int i1 = 0; float b1v = w0;
    if (w1 > b1v) { b1v = w1; i1 = 1; }
    if (w2 > b1v) { b1v = w2; i1 = 2; }
    if (w3 > b1v) { b1v = w3; i1 = 3; }
    int i2 = -1; float b2v = -1.0f;
    if (i1 != 0 && w0 > b2v) { b2v = w0; i2 = 0; }
    if (i1 != 1 && w1 > b2v) { b2v = w1; i2 = 1; }
    if (i1 != 2 && w2 > b2v) { b2v = w2; i2 = 2; }
    if (i1 != 3 && w3 > b2v) { b2v = w3; i2 = 3; }
    float d0 = (i1 == 0 || i2 == 0) ? w0 : 0.0f;
    float d1 = (i1 == 1 || i2 == 1) ? w1 : 0.0f;
    float d2 = (i1 == 2 || i2 == 2) ? w2 : 0.0f;
    float d3 = (i1 == 3 || i2 == 3) ? w3 : 0.0f;

    if (hi == 0) {   // lanes 0-15 own the per-row router outputs
      ((float4*)raw_out)[row] = make_float4(w0, w1, w2, w3);
      ((float4*)msk_out)[row] = make_float4(d0 > 0.f || i1 == 0 || i2 == 0 ? 1.f : 0.f,
                                            i1 == 1 || i2 == 1 ? 1.f : 0.f,
                                            i1 == 2 || i2 == 2 ? 1.f : 0.f,
                                            i1 == 3 || i2 == 3 ? 1.f : 0.f);
    }

    // ---- expert 0: fourier (K=32), dispatch weight folded into A ----
    {
      v16h a;
#pragma unroll
      for (int j = 0; j < 8; ++j) {
        float arg = tv * pf_freq[j];
        a[j]     = (_Float16)(d0 * __sinf(arg));   // K = hi*8+j      (sin block)
        a[j + 8] = (_Float16)(d0 * __cosf(arg));   // K = 16+hi*8+j   (cos block)
      }
      run_expert(a, 0, lane, ldsB, rowBase, emb);
    }
    // ---- expert 1: spline RBF-on-grid (K=16, upper half zero) ----
    {
      v16h a;
#pragma unroll
      for (int j = 0; j < 8; ++j) {
        float z = (tv - pf_knot[j]) * inv_h;
        a[j]     = (_Float16)(d1 * __expf(-z * z));
        a[j + 8] = (_Float16)0.0f;
      }
      run_expert(a, 1, lane, ldsB, rowBase, emb);
    }
    // ---- expert 2: rkhs gaussian ----
    {
      v16h a;
#pragma unroll
      for (int j = 0; j < 8; ++j) {
        float z = tv - pf_cent[j];
        a[j]     = (_Float16)(d2 * __expf(-z * z));
        a[j + 8] = (_Float16)0.0f;
      }
      run_expert(a, 2, lane, ldsB, rowBase, emb);
    }
    // ---- expert 3: mexican-hat wavelet ----
    {
      v16h a;
#pragma unroll
      for (int j = 0; j < 8; ++j) {
        float u = (tv - pf_wc[j]) * pf_wsi[j];
        a[j]     = (_Float16)(d3 * ((1.0f - u * u) * __expf(-0.5f * u * u)));
        a[j + 8] = (_Float16)0.0f;
      }
      run_expert(a, 3, lane, ldsB, rowBase, emb);
    }
  }
}

extern "C" void kernel_launch(void* const* d_in, const int* in_sizes, int n_in,
                              void* d_out, int out_size, void* d_ws, size_t ws_size,
                              hipStream_t stream) {
  const int B = in_sizes[0];                       // timestamp is [B,1]
  const float* ts        = (const float*)d_in[0];
  const float* aux       = (const float*)d_in[1];
  const float* router_W  = (const float*)d_in[2];
  const float* router_b  = (const float*)d_in[3];
  const float* freqs     = (const float*)d_in[4];
  const float* fourier_W = (const float*)d_in[5];
  const float* knots     = (const float*)d_in[6];
  const float* spline_W  = (const float*)d_in[7];
  const float* centers   = (const float*)d_in[8];
  const float* gauss_W   = (const float*)d_in[9];
  const float* wav_c     = (const float*)d_in[10];
  const float* wav_s     = (const float*)d_in[11];
  const float* wavelet_W = (const float*)d_in[12];

  float* emb = (float*)d_out;                      // [B, 512]
  float* raw = emb + (size_t)B * 512;              // [B, 4]
  float* msk = raw + (size_t)B * 4;                // [B, 4] (0.0/1.0)

  mote_moe_kernel<<<dim3(NBLOCKS), dim3(THREADS), 0, stream>>>(
      ts, aux, router_W, router_b, freqs, fourier_W, knots, spline_W,
      centers, gauss_W, wav_c, wav_s, wavelet_W, emb, raw, msk, B);
}
